// CrossAttention_77240691851613
// MI455X (gfx1250) — matrix operand order repaired and checked by reading, not compiled, not run on hardware
//
#include <hip/hip_runtime.h>
#include <hip/hip_bf16.h>

typedef __attribute__((ext_vector_type(16))) _Float16 v16h;
typedef __attribute__((ext_vector_type(8)))  float    v8f;
typedef __attribute__((ext_vector_type(4)))  unsigned v4u;
typedef __attribute__((ext_vector_type(8)))  int      v8i;
typedef __attribute__((ext_vector_type(4)))  int      v4i;

#if defined(__has_builtin)
#if __has_builtin(__builtin_amdgcn_tensor_load_to_lds) && \
    __has_builtin(__builtin_amdgcn_s_wait_tensorcnt)
#define HAVE_TDM 1
#endif
#endif

union V16H {
    v16h v;
    _Float16 h[16];
    unsigned u[8];
};

static __device__ __forceinline__ unsigned pack2f16(float a, float b) {
    union { _Float16 h[2]; unsigned u; } p;
    p.h[0] = (_Float16)a;
    p.h[1] = (_Float16)b;
    return p.u;
}

static __device__ __forceinline__ unsigned short f32_to_f16_bits(float a) {
    union { _Float16 h; unsigned short s; } p;
    p.h = (_Float16)a;
    return p.s;
}

#ifdef HAVE_TDM
// ---------------------------------------------------------------------------
// Tensor Data Mover: DMA a 2-D tile (rows x rowlen of 2-byte elements) from
// global memory into LDS.  D# per CDNA5 ISA ch.8:
//   group0: count=1 | lds_addr | global_addr(57b) | type=2
//   group1: data_size=1(2B), tensor_dim0, tensor_dim1, tile_dim0/1, dim0_stride
//   groups 2/3 (+ extra group on this toolchain): zero (2-D tensor)
// Tracked by TENSORcnt; in-order per wave.
// This toolchain's builtin arity is 6: (v4u, v8i, v4i, v4i, v8i, i32 cpol).
// ---------------------------------------------------------------------------
static __device__ __forceinline__ void tdm_load_tile_f16(
        unsigned lds_off, const void* gptr,
        int rows, int rowlen, int row_stride, int tensor_rows) {
    const unsigned long long ga = (unsigned long long)(uintptr_t)gptr;
    v4u g0;
    g0[0] = 1u;                                           // count=1, user mode
    g0[1] = lds_off;                                      // lds_addr
    g0[2] = (unsigned)(ga & 0xffffffffu);                 // global_addr[31:0]
    g0[3] = (unsigned)((ga >> 32) & 0x01ffffffu)          // global_addr[56:32]
          | (2u << 30);                                   // type=2 ("image")
    v8i g1;
    g1[0] = (1 << 16);                                    // data_size=2B, mask=0
    g1[1] = (rowlen & 0xffff) << 16;                      // tensor_dim0[15:0]
    g1[2] = (tensor_rows & 0xffff) << 16;                 // tensor_dim1[15:0]
    g1[3] = ((rowlen & 0xffff) << 16)                     // tile_dim0
          | ((tensor_rows >> 16) & 0xffff);               // tensor_dim1[31:16]
    g1[4] = rows & 0xffff;                                // tile_dim1 (tile_dim2=0)
    g1[5] = row_stride;                                   // tensor_dim0_stride[31:0]
    g1[6] = 0;
    g1[7] = 0;
    const v4i z4 = {0, 0, 0, 0};
    const v8i z8 = {0, 0, 0, 0, 0, 0, 0, 0};
    __builtin_amdgcn_tensor_load_to_lds(g0, g1, z4, z4, z8, 0);
}
#endif

// ---------------------------------------------------------------------------
// Kernel 1: projection GEMM  C[4096,512] = X[4096,64] @ W[64,512] + bias
// writes f16 result rearranged per-head: dst[((b*8+h)*2048 + l)*64 + d]
// 8192 waves (256 row-tiles x 32 col-tiles), 4 waves / 128-thread block.
// ---------------------------------------------------------------------------
__global__ void proj_kernel(const float* __restrict__ X,
                            const float* __restrict__ W,
                            const float* __restrict__ bias,
                            unsigned short* __restrict__ dst) {
    const int wave = blockIdx.x * (blockDim.x >> 5) + (threadIdx.x >> 5);
    const int lane = threadIdx.x & 31;
    const int g = lane >> 4;
    const int mr = lane & 15;
    const int ct = wave & 31;
    const int rt = wave >> 5;

    v8f acc = {};
    const float* xrow = X + (size_t)(rt * 16 + mr) * 64;
    const int n = ct * 16 + mr;

#pragma unroll
    for (int ks = 0; ks < 64; ks += 32) {
        V16H A, B;
#pragma unroll
        for (int v = 0; v < 8; ++v) {
            const int ka = ks + (g << 4) + (v << 1);
            const float2 f = *(const float2*)(xrow + ka);
            A.u[v] = pack2f16(f.x, f.y);
            const int kb = ks + (g << 4) + (v << 1);
            B.u[v] = pack2f16(W[(size_t)kb * 512 + n], W[(size_t)(kb + 1) * 512 + n]);
        }
        acc = __builtin_amdgcn_wmma_f32_16x16x32_f16(false, A.v, false, B.v,
                                                     (short)0, acc, false, false);
    }

    const float bn = bias[n];
    const int h = n >> 6;
    const int d = n & 63;
#pragma unroll
    for (int r = 0; r < 8; ++r) {
        const int row = rt * 16 + r + (g << 3);
        const int b = row >> 11;
        const int l = row & 2047;
        dst[(size_t)(((b << 3) + h) * 2048 + l) * 64 + d] = f32_to_f16_bits(acc[r] + bn);
    }
}

// ---------------------------------------------------------------------------
// Kernel 2: flash attention per (b,h).  attn[k,l] = sum_d K[k,d]*Q[l,d] / 8,
// softmax over l (streamed), O[k,d] = sum_l P[k,l]*V[l,d].
// One wave = 16 rows of K, loops over l in chunks of 32.  With TDM, each wave
// double-buffers the Q and V chunks in private LDS via tensor_load_to_lds
// (pipelined: DMA chunk i+1 overlaps compute of chunk i; s_wait_tensorcnt 2).
// 2048 waves = 16 (b,h) x 128 row-tiles; 512 blocks of 128 threads.
// ---------------------------------------------------------------------------
__global__ void attn_kernel(const unsigned short* __restrict__ Q,
                            const unsigned short* __restrict__ K,
                            const unsigned short* __restrict__ V,
                            float* __restrict__ out) {
    __shared__ unsigned short Plds[4][16 * 40];   // per-wave padded P scratch
#ifdef HAVE_TDM
    __shared__ unsigned short Qch[4][2][32 * 64]; // per-wave double-buffered
    __shared__ unsigned short Vch[4][2][32 * 64]; // chunk staging (TDM dest)
#endif

    const int w = threadIdx.x >> 5;
    const int lane = threadIdx.x & 31;
    const int g = lane >> 4;
    const int mr = lane & 15;
    const int wave = blockIdx.x * 4 + w;
    const int rt = wave & 127;
    const int bh = wave >> 7;

    const unsigned short* Qb = Q + (size_t)bh * 2048 * 64;
    const unsigned short* Kb = K + (size_t)bh * 2048 * 64;
    const unsigned short* Vb = V + (size_t)bh * 2048 * 64;

    // Stationary A-tiles: 16 K-rows, d split 0..31 / 32..63
    V16H AK0, AK1;
    {
        const unsigned short* krow = Kb + (size_t)(rt * 16 + mr) * 64;
#pragma unroll
        for (int v = 0; v < 8; ++v) {
            const int ka = (g << 4) + (v << 1);
            AK0.u[v] = *(const unsigned*)(krow + ka);
            AK1.u[v] = *(const unsigned*)(krow + 32 + ka);
        }
    }

    float m[8], s[8];
    v8f O[4] = {};
#pragma unroll
    for (int r = 0; r < 8; ++r) { m[r] = -3.0e38f; s[r] = 0.0f; }

#ifdef HAVE_TDM
    const unsigned qoff0 = (unsigned)(uintptr_t)&Qch[w][0][0];
    const unsigned qoff1 = (unsigned)(uintptr_t)&Qch[w][1][0];
    const unsigned voff0 = (unsigned)(uintptr_t)&Vch[w][0][0];
    const unsigned voff1 = (unsigned)(uintptr_t)&Vch[w][1][0];
    // Prologue: DMA chunk 0 into buffer 0.
    tdm_load_tile_f16(qoff0, Qb, 32, 64, 64, 2048);
    tdm_load_tile_f16(voff0, Vb, 32, 64, 64, 2048);
#endif

    for (int lc = 0; lc < 64; ++lc) {
        const int l0 = lc * 32;

#ifdef HAVE_TDM
        const int buf = lc & 1;
        if (lc + 1 < 64) {
            // Issue DMA for next chunk, then wait for current pair only
            // (TDM ops complete in-order per wave -> cnt<=2 means chunk lc done).
            tdm_load_tile_f16(buf ? qoff0 : qoff1,
                              Qb + (size_t)(l0 + 32) * 64, 32, 64, 64, 2048);
            tdm_load_tile_f16(buf ? voff0 : voff1,
                              Vb + (size_t)(l0 + 32) * 64, 32, 64, 64, 2048);
            __builtin_amdgcn_s_wait_tensorcnt(2);
        } else {
            __builtin_amdgcn_s_wait_tensorcnt(0);
        }
        const unsigned short* Ql = &Qch[w][buf][0];
        const unsigned short* Vl = &Vch[w][buf][0];
        // Escape the staging pointers: TDM writes are invisible to alias
        // analysis; this stops the compiler reasoning "never stored".
        asm volatile("" ::"v"((unsigned)(uintptr_t)Ql),
                         "v"((unsigned)(uintptr_t)Vl)
                     : "memory");
#else
        if (lc + 1 < 64) {
            __builtin_prefetch(Vb + (size_t)(l0 + 32) * 64, 0, 0);
            __builtin_prefetch(Qb + (size_t)(l0 + 32) * 64, 0, 0);
        }
#endif

        // ---- S = (K_rows @ Q^T) for 32 columns (two 16-wide halves) ----
        v8f S[2];
#pragma unroll
        for (int half = 0; half < 2; ++half) {
            V16H B0, B1;
#ifdef HAVE_TDM
            const unsigned short* qrow = Ql + (size_t)(half * 16 + mr) * 64;
#else
            const unsigned short* qrow = Qb + (size_t)(l0 + half * 16 + mr) * 64;
#endif
#pragma unroll
            for (int v = 0; v < 8; ++v) {
                const int kb = (g << 4) + (v << 1);      // K index = d
                B0.u[v] = *(const unsigned*)(qrow + kb);
                B1.u[v] = *(const unsigned*)(qrow + 32 + kb);
            }
            v8f t = {};
            t = __builtin_amdgcn_wmma_f32_16x16x32_f16(false, AK0.v, false, B0.v,
                                                       (short)0, t, false, false);
            t = __builtin_amdgcn_wmma_f32_16x16x32_f16(false, AK1.v, false, B1.v,
                                                       (short)0, t, false, false);
            S[half] = t;
        }

        // ---- streaming softmax over this 32-wide chunk ----
        float alpha[8];
#pragma unroll
        for (int r = 0; r < 8; ++r) {
            float a0 = S[0][r] * 0.125f;
            float a1 = S[1][r] * 0.125f;
            S[0][r] = a0;
            S[1][r] = a1;
            float cmax = fmaxf(a0, a1);
            cmax = fmaxf(cmax, __shfl_xor(cmax, 1));
            cmax = fmaxf(cmax, __shfl_xor(cmax, 2));
            cmax = fmaxf(cmax, __shfl_xor(cmax, 4));
            cmax = fmaxf(cmax, __shfl_xor(cmax, 8));
            const float mn = fmaxf(m[r], cmax);
            alpha[r] = __expf(m[r] - mn);
            m[r] = mn;
            const float p0 = __expf(S[0][r] - mn);
            const float p1 = __expf(S[1][r] - mn);
            S[0][r] = p0;
            S[1][r] = p1;
            float prow = p0 + p1;
            prow += __shfl_xor(prow, 1);
            prow += __shfl_xor(prow, 2);
            prow += __shfl_xor(prow, 4);
            prow += __shfl_xor(prow, 8);
            s[r] = s[r] * alpha[r] + prow;
        }
#pragma unroll
        for (int r = 0; r < 8; ++r) {
            const float a = alpha[r];
            O[0][r] *= a; O[1][r] *= a; O[2][r] *= a; O[3][r] *= a;
        }

        // ---- remap P from C-layout to A-layout via per-wave LDS ----
#pragma unroll
        for (int half = 0; half < 2; ++half) {
#pragma unroll
            for (int r = 0; r < 8; ++r) {
                const int row = r + (g << 3);
                const int col = half * 16 + mr;
                Plds[w][row * 40 + col] = f32_to_f16_bits(S[half][r]);
            }
        }
        V16H AP;
#pragma unroll
        for (int v = 0; v < 8; ++v) {
            const int ka = (g << 4) + (v << 1);
            AP.u[v] = *(const unsigned*)&Plds[w][mr * 40 + ka];
        }

        // ---- O[16,64] += P[16,32] @ V[32,64]  (4 N-tiles of 16) ----
#pragma unroll
        for (int t = 0; t < 4; ++t) {
            V16H BV;
            const int d = t * 16 + mr;
#pragma unroll
            for (int v = 0; v < 8; ++v) {
                const int kll = (g << 4) + (v << 1);     // K index = l (chunk-local)
#ifdef HAVE_TDM
                const unsigned lo = Vl[(size_t)kll * 64 + d];
                const unsigned hi = Vl[(size_t)(kll + 1) * 64 + d];
#else
                const unsigned lo = Vb[(size_t)(l0 + kll) * 64 + d];
                const unsigned hi = Vb[(size_t)(l0 + kll + 1) * 64 + d];
#endif
                BV.u[v] = lo | (hi << 16);
            }
            O[t] = __builtin_amdgcn_wmma_f32_16x16x32_f16(false, AP.v, false, BV.v,
                                                          (short)0, O[t], false, false);
        }
    }

    // ---- epilogue: normalize and store per-head output [b,h,lk,d] f32 ----
#pragma unroll
    for (int t = 0; t < 4; ++t) {
#pragma unroll
        for (int r = 0; r < 8; ++r) {
            const int lk = rt * 16 + r + (g << 3);
            out[(size_t)(bh * 2048 + lk) * 64 + t * 16 + mr] = O[t][r] / s[r];
        }
    }
}

// ---------------------------------------------------------------------------
// Kernel 3: mean over heads, write transposed f16:  meanT[b, d, l]
// ---------------------------------------------------------------------------
__global__ void mean_kernel(const float* __restrict__ attn,
                            unsigned short* __restrict__ meanT) {
    const int idx = blockIdx.x * blockDim.x + threadIdx.x;
    if (idx >= 2 * 2048 * 64) return;
    const int d = idx & 63;
    const int l = (idx >> 6) & 2047;
    const int b = idx >> 17;
    float acc = 0.0f;
#pragma unroll
    for (int h = 0; h < 8; ++h)
        acc += attn[(size_t)(((b << 3) + h) * 2048 + l) * 64 + d];
    acc *= 0.125f;
    meanT[(size_t)(b * 64 + d) * 2048 + l] = f32_to_f16_bits(acc);
}

// ---------------------------------------------------------------------------
// Kernel 4: final GEMM  out[b, d, n] = meanT[b, d, :] @ Wl[:, n] + bl[n]
// 360 waves = 2 b x 4 row-tiles x 45 col-tiles; 90 blocks of 128 threads.
// ---------------------------------------------------------------------------
__global__ void final_kernel(const unsigned short* __restrict__ meanT,
                             const float* __restrict__ Wl,
                             const float* __restrict__ bl,
                             float* __restrict__ out) {
    const int wave = blockIdx.x * (blockDim.x >> 5) + (threadIdx.x >> 5);
    const int lane = threadIdx.x & 31;
    const int g = lane >> 4;
    const int mr = lane & 15;
    const int nt = wave % 45;
    const int mt = (wave / 45) & 3;
    const int b = wave / 180;

    const unsigned short* arow = meanT + (size_t)(b * 64 + mt * 16 + mr) * 2048;
    const int n = nt * 16 + mr;
    v8f acc = {};

    for (int kc = 0; kc < 64; ++kc) {
        const int ks = kc * 32;
        V16H A, B;
#pragma unroll
        for (int v = 0; v < 8; ++v) {
            const int ka = ks + (g << 4) + (v << 1);
            A.u[v] = *(const unsigned*)(arow + ka);
            const int kb = ks + (g << 4) + (v << 1);
            B.u[v] = pack2f16(Wl[(size_t)kb * 720 + n], Wl[(size_t)(kb + 1) * 720 + n]);
        }
        acc = __builtin_amdgcn_wmma_f32_16x16x32_f16(false, A.v, false, B.v,
                                                     (short)0, acc, false, false);
    }

    const float bn = bl[n];
#pragma unroll
    for (int r = 0; r < 8; ++r) {
        const int drow = mt * 16 + r + (g << 3);
        out[(size_t)b * 64 * 720 + (size_t)drow * 720 + n] = acc[r] + bn;
    }
}

// ---------------------------------------------------------------------------
extern "C" void kernel_launch(void* const* d_in, const int* in_sizes, int n_in,
                              void* d_out, int out_size, void* d_ws, size_t ws_size,
                              hipStream_t stream) {
    const float* query = (const float*)d_in[0];
    const float* key   = (const float*)d_in[1];
    const float* value = (const float*)d_in[2];
    const float* Wq = (const float*)d_in[3];
    const float* bq = (const float*)d_in[4];
    const float* Wk = (const float*)d_in[5];
    const float* bk = (const float*)d_in[6];
    const float* Wv = (const float*)d_in[7];
    const float* bv = (const float*)d_in[8];
    const float* Wl = (const float*)d_in[9];
    const float* bl = (const float*)d_in[10];

    const size_t QKV_ELEMS = (size_t)2 * 8 * 2048 * 64;

    unsigned short* qbuf = (unsigned short*)d_ws;
    unsigned short* kbuf = qbuf + QKV_ELEMS;
    unsigned short* vbuf = kbuf + QKV_ELEMS;
    float* attnout = (float*)(vbuf + QKV_ELEMS);
    unsigned short* meanT = (unsigned short*)(attnout + QKV_ELEMS);

    proj_kernel<<<2048, 128, 0, stream>>>(query, Wq, bq, qbuf);
    proj_kernel<<<2048, 128, 0, stream>>>(key,   Wk, bk, kbuf);
    proj_kernel<<<2048, 128, 0, stream>>>(value, Wv, bv, vbuf);

    attn_kernel<<<512, 128, 0, stream>>>(qbuf, kbuf, vbuf, attnout);

    mean_kernel<<<1024, 256, 0, stream>>>(attnout, meanT);

    final_kernel<<<90, 128, 0, stream>>>(meanT, Wl, bl, (float*)d_out);
}
